// GAT_2_36464272343053
// MI455X (gfx1250) — compile-verified, matrix operand order checked by reading, not run on hardware
//
#include <hip/hip_runtime.h>
#include <hip/hip_bf16.h>

#define NNODE  4096
#define NFEAT  512
#define NHID   64
#define NCLASS 40
#define NHEADS 8
#define ALPHA  0.2f
#define LOG2E  1.4426950408889634f

typedef __attribute__((ext_vector_type(8)))  float  v8f;
typedef __attribute__((ext_vector_type(16))) __bf16 v16bf;

union Frag { v16bf v; unsigned int u[8]; };

static __device__ __forceinline__ unsigned short f2bf(float f) {
    unsigned int u = __float_as_uint(f);
    unsigned int r = (u + 0x7FFFu + ((u >> 16) & 1u)) >> 16;   // RNE
    return (unsigned short)r;
}
static __device__ __forceinline__ float bf2f(unsigned short h) {
    return __uint_as_float(((unsigned int)h) << 16);
}
static __device__ __forceinline__ unsigned int pack2(unsigned short lo, unsigned short hi) {
    return (unsigned int)lo | ((unsigned int)hi << 16);
}
// truncating bf16 pack: 2 VALU ops, fine for probabilities in [0,1]
static __device__ __forceinline__ unsigned int pack2t(float lo, float hi) {
    return (__float_as_uint(lo) >> 16) | (__float_as_uint(hi) & 0xFFFF0000u);
}
// leaky relu as max(s, alpha*s): valid for alpha in (0,1)
static __device__ __forceinline__ float lrelu(float s) {
    return fmaxf(s, ALPHA * s);
}
static __device__ __forceinline__ float exp2_(float x) {
#if __has_builtin(__builtin_amdgcn_exp2f)
    return __builtin_amdgcn_exp2f(x);          // bare v_exp_f32
#else
    return exp2f(x);
#endif
}

// -------------------------------------------------------------------------
// k1: Wh[h] = x @ W1[h]  (bf16 WMMA, f32 accum), stored transposed:
//     whT[h][d][node] bf16.  grid(256,4,8), block 32 (one wave).
// -------------------------------------------------------------------------
__global__ __launch_bounds__(32) void k1_wh(const float* __restrict__ x,
                                            const float* __restrict__ W1,
                                            unsigned short* __restrict__ whT) {
    const int l = threadIdx.x, m = l & 15, half = l >> 4;
    const int rowbase = blockIdx.x * 16;
    const int colbase = blockIdx.y * 16;
    const int h = blockIdx.z;
    v8f c = {};
    for (int k0 = 0; k0 < NFEAT; k0 += 32) {
        Frag a, b;
#pragma unroll
        for (int v = 0; v < 8; ++v) {
            const int k = k0 + ((v >> 2) << 4) + (half << 3) + ((v & 3) << 1);
            const float2 xv = *(const float2*)(x + (size_t)(rowbase + m) * NFEAT + k);
            a.u[v] = pack2(f2bf(xv.x), f2bf(xv.y));
            const float* wp = W1 + ((size_t)h * NFEAT + k) * NHID + colbase + m;
            b.u[v] = pack2(f2bf(wp[0]), f2bf(wp[NHID]));
        }
        c = __builtin_amdgcn_wmma_f32_16x16x32_bf16(false, a.v, false, b.v,
                                                    (short)0, c, false, false);
    }
#pragma unroll
    for (int r = 0; r < 8; ++r) {
        const int row = rowbase + half * 8 + r;
        whT[((size_t)h * NHID + (colbase + m)) * NNODE + row] = f2bf(c[r]);
    }
}

// -------------------------------------------------------------------------
// k2: f1[h,n] = log2(e) * Wh[h,n,:]·a1[h,0,:]  (log2-domain logits)
// -------------------------------------------------------------------------
__global__ __launch_bounds__(256) void k2_f1f2(const unsigned short* __restrict__ whT,
                                               const float* __restrict__ a1,
                                               float* __restrict__ f1,
                                               float* __restrict__ f2) {
    const int n = blockIdx.x * 256 + threadIdx.x;
    const int h = blockIdx.y;
    float s1 = 0.f, s2 = 0.f;
    for (int d = 0; d < NHID; ++d) {
        const float w = bf2f(whT[((size_t)h * NHID + d) * NNODE + n]);
        s1 += w * a1[((size_t)h * 2 + 0) * NHID + d];
        s2 += w * a1[((size_t)h * 2 + 1) * NHID + d];
    }
    f1[(size_t)h * NNODE + n] = s1 * LOG2E;
    f2[(size_t)h * NNODE + n] = s2 * LOG2E;
}

// -------------------------------------------------------------------------
// k3: fused attention layer 1.  8 waves/block (one head each).  Per 32-col
//     step each wave ballots 2 adj rows into 32-bit masks shared via a
//     ping-pong LDS buffer -> ONE barrier per step, and next-tile adj loads
//     are issued right after the barrier so latency hides under compute.
//     Unstabilized softmax (logits bounded ~6; exp cannot overflow).
//     4x v_wmma_f32_16x16x32_bf16 per step.  Writes ELU'd h1 bf16.
// -------------------------------------------------------------------------
__global__ __launch_bounds__(256) void k3_attn1(const int* __restrict__ adj1,
                                                const unsigned short* __restrict__ whT,
                                                const float* __restrict__ f1,
                                                const float* __restrict__ f2,
                                                unsigned short* __restrict__ h1b) {
    __shared__ unsigned int maskS[2][16];   // ping-pong row masks
    const int tid = threadIdx.x;
    const int l = tid & 31, h = tid >> 5;
    const int m = l & 15, half = l >> 4;
    const int rb = blockIdx.x;
    const float f1val = f1[(size_t)h * NNODE + rb * 16 + m];   // log2 domain

    float Sr = 0.f;           // per-lane partial row sum (combined at end)
    v8f acc[4] = {};

    // this wave owns adjacency rows 2h and 2h+1; lane l covers one column
    const int* aprow = adj1 + (size_t)(rb * 16 + 2 * h) * NNODE + l;
    int av0 = aprow[0];
    int av1 = aprow[NNODE];

    for (int j = 0; j < NNODE / 32; ++j) {
        const int colbase = j * 32;
        const int buf = j & 1;
        const unsigned int mk0 = (unsigned int)__ballot(av0 != 0);
        const unsigned int mk1 = (unsigned int)__ballot(av1 != 0);
        if (l == 0) { maskS[buf][2 * h] = mk0; maskS[buf][2 * h + 1] = mk1; }
        __syncthreads();

        if (j + 1 < NNODE / 32) {          // issue next tile's loads early
            av0 = aprow[colbase + 32];
            av1 = aprow[NNODE + colbase + 32];
            if (j + 2 < NNODE / 32) {      // keep HBM stream 2 tiles ahead
                __builtin_prefetch(aprow + colbase + 64, 0, 1);
                __builtin_prefetch(aprow + NNODE + colbase + 64, 0, 1);
            }
        }
        const unsigned int rmask = maskS[buf][m];

        float psum = 0.f;
        Frag a;
#pragma unroll
        for (int v = 0; v < 8; ++v) {
            const int k = ((v >> 2) << 4) + (half << 3) + ((v & 3) << 1);
            const float2 fv = *(const float2*)(f2 + (size_t)h * NNODE + colbase + k);
            float p0 = exp2_(lrelu(f1val + fv.x));
            float p1 = exp2_(lrelu(f1val + fv.y));
            if (!((rmask >> k) & 1u))       p0 = 0.f;
            if (!((rmask >> (k + 1)) & 1u)) p1 = 0.f;
            psum += p0 + p1;
            a.u[v] = pack2t(p0, p1);
        }
        Sr += psum;

        const unsigned short* wb = whT + (size_t)h * NHID * NNODE + colbase;
#pragma unroll
        for (int g = 0; g < 4; ++g) {
            Frag b;
#pragma unroll
            for (int v = 0; v < 8; ++v) {
                const int k = ((v >> 2) << 4) + (half << 3) + ((v & 3) << 1);
                b.u[v] = *(const unsigned int*)(wb + (size_t)(g * 16 + m) * NNODE + k);
            }
            acc[g] = __builtin_amdgcn_wmma_f32_16x16x32_bf16(false, a.v, false, b.v,
                                                             (short)0, acc[g], false, false);
        }
    }

    Sr += __shfl_xor(Sr, 16, 32);   // combine the two K-halves of each row
#pragma unroll
    for (int r = 0; r < 8; ++r) {
        const float sr = __shfl(Sr, half * 8 + r, 32);
        const float inv = 1.f / sr;
        const int row = rb * 16 + half * 8 + r;
        unsigned short* op = h1b + (size_t)row * (NHEADS * NHID) + h * NHID;
#pragma unroll
        for (int g = 0; g < 4; ++g) {
            float v0 = acc[g][r] * inv;
            v0 = v0 > 0.f ? v0 : __expf(v0) - 1.f;   // ELU (concat=True layer)
            op[g * 16 + m] = f2bf(v0);
        }
    }
}

// -------------------------------------------------------------------------
// k4: Wh2 = h1 @ Wo  (40 classes padded to 48), stored transposed bf16
//     wh2T[48][4096] (pad rows exact zeros).  grid(256,3), block 32.
// -------------------------------------------------------------------------
__global__ __launch_bounds__(32) void k4_wh2(const unsigned short* __restrict__ h1b,
                                             const float* __restrict__ Wo,
                                             unsigned short* __restrict__ wh2T) {
    const int l = threadIdx.x, m = l & 15, half = l >> 4;
    const int rowbase = blockIdx.x * 16;
    const int colbase = blockIdx.y * 16;
    const int col = colbase + m;
    v8f c = {};
    for (int k0 = 0; k0 < NHEADS * NHID; k0 += 32) {
        Frag a, b;
#pragma unroll
        for (int v = 0; v < 8; ++v) {
            const int k = k0 + ((v >> 2) << 4) + (half << 3) + ((v & 3) << 1);
            a.u[v] = *(const unsigned int*)(h1b + (size_t)(rowbase + m) * (NHEADS * NHID) + k);
            const float w0 = (col < NCLASS) ? Wo[(size_t)k * NCLASS + col] : 0.f;
            const float w1 = (col < NCLASS) ? Wo[(size_t)(k + 1) * NCLASS + col] : 0.f;
            b.u[v] = pack2(f2bf(w0), f2bf(w1));
        }
        c = __builtin_amdgcn_wmma_f32_16x16x32_bf16(false, a.v, false, b.v,
                                                    (short)0, c, false, false);
    }
#pragma unroll
    for (int r = 0; r < 8; ++r)
        wh2T[(size_t)col * NNODE + rowbase + half * 8 + r] = f2bf(c[r]);
}

// -------------------------------------------------------------------------
// k5: layer-2 logits f12/f22 (log2 domain)
// -------------------------------------------------------------------------
__global__ __launch_bounds__(256) void k5_f1f2(const unsigned short* __restrict__ wh2T,
                                               const float* __restrict__ ao,
                                               float* __restrict__ f12,
                                               float* __restrict__ f22) {
    const int n = blockIdx.x * 256 + threadIdx.x;
    float s1 = 0.f, s2 = 0.f;
    for (int d = 0; d < NCLASS; ++d) {
        const float w = bf2f(wh2T[(size_t)d * NNODE + n]);
        s1 += w * ao[d];
        s2 += w * ao[NCLASS + d];
    }
    f12[n] = s1 * LOG2E;
    f22[n] = s2 * LOG2E;
}

// -------------------------------------------------------------------------
// k6: fused attention layer 2 + outer ELU + log_softmax.  8 waves/block;
//     each wave reduces a 512-column slab (partial acc + partial row sums),
//     partials combined through LDS (attention is linear in them), then
//     2 rows per wave are finalized with a stabilized class log_softmax.
// -------------------------------------------------------------------------
__global__ __launch_bounds__(256) void k6_attn2(const int* __restrict__ adj0,
                                                const unsigned short* __restrict__ wh2T,
                                                const float* __restrict__ f12,
                                                const float* __restrict__ f22,
                                                float* __restrict__ out) {
    __shared__ float accS[8][16][48];   // per-wave partial output tiles (24 KB)
    __shared__ float SrS[8][16];        // per-wave partial row sums
    const int tid = threadIdx.x;
    const int l = tid & 31, w = tid >> 5;
    const int m = l & 15, half = l >> 4;
    const int rb = blockIdx.x;
    const float f1val = f12[rb * 16 + m];   // log2 domain

    float Sr = 0.f;
    v8f acc[3] = {};

    const int jbeg = w * (NNODE / 32 / 8);
    const int jend = jbeg + (NNODE / 32 / 8);
    for (int j = jbeg; j < jend; ++j) {
        const int colbase = j * 32;
        const int* ap = adj0 + (size_t)(rb * 16 + m) * NNODE + colbase;
        if (j + 1 < jend) __builtin_prefetch(ap + 32, 0, 1);

        float psum = 0.f;
        Frag a;
#pragma unroll
        for (int v = 0; v < 8; ++v) {
            const int k = ((v >> 2) << 4) + (half << 3) + ((v & 3) << 1);
            const float2 fv = *(const float2*)(f22 + colbase + k);
            const int2 av = *(const int2*)(ap + k);
            float p0 = exp2_(lrelu(f1val + fv.x));
            float p1 = exp2_(lrelu(f1val + fv.y));
            if (av.x == 0) p0 = 0.f;
            if (av.y == 0) p1 = 0.f;
            psum += p0 + p1;
            a.u[v] = pack2t(p0, p1);
        }
        Sr += psum;

#pragma unroll
        for (int g = 0; g < 3; ++g) {
            Frag b;
#pragma unroll
            for (int v = 0; v < 8; ++v) {
                const int k = ((v >> 2) << 4) + (half << 3) + ((v & 3) << 1);
                b.u[v] = *(const unsigned int*)(wh2T + (size_t)(g * 16 + m) * NNODE + colbase + k);
            }
            acc[g] = __builtin_amdgcn_wmma_f32_16x16x32_bf16(false, a.v, false, b.v,
                                                             (short)0, acc[g], false, false);
        }
    }

    // stash this wave's partials
    Sr += __shfl_xor(Sr, 16, 32);
    if (l < 16) SrS[w][m] = Sr;
#pragma unroll
    for (int g = 0; g < 3; ++g)
#pragma unroll
        for (int r = 0; r < 8; ++r)
            accS[w][half * 8 + r][g * 16 + m] = acc[g][r];
    __syncthreads();

    // finalize: wave w handles rows 2w (lanes 0-15) and 2w+1 (lanes 16-31)
    const int row16 = 2 * w + half;
    float Srow = 0.f;
#pragma unroll
    for (int q = 0; q < 8; ++q) Srow += SrS[q][row16];
    const float inv = 1.f / Srow;

    float e[3];
#pragma unroll
    for (int g = 0; g < 3; ++g) {
        float s = 0.f;
#pragma unroll
        for (int q = 0; q < 8; ++q) s += accS[q][row16][g * 16 + m];
        s *= inv;                                  // attention output
        e[g] = s > 0.f ? s : __expf(s) - 1.f;      // outer ELU
    }
    // log_softmax over 40 valid classes: col = g*16+m, g==2 valid iff m<8
    float mx = fmaxf(e[0], e[1]);
    if (m < 8) mx = fmaxf(mx, e[2]);
#pragma unroll
    for (int off = 1; off < 16; off <<= 1) mx = fmaxf(mx, __shfl_xor(mx, off, 32));
    float se = __expf(e[0] - mx) + __expf(e[1] - mx);
    if (m < 8) se += __expf(e[2] - mx);
#pragma unroll
    for (int off = 1; off < 16; off <<= 1) se += __shfl_xor(se, off, 32);
    const float lse = __logf(se);

    const int row = rb * 16 + row16;
    out[(size_t)row * NCLASS + m]      = e[0] - mx - lse;
    out[(size_t)row * NCLASS + 16 + m] = e[1] - mx - lse;
    if (m < 8) out[(size_t)row * NCLASS + 32 + m] = e[2] - mx - lse;
}

// -------------------------------------------------------------------------
extern "C" void kernel_launch(void* const* d_in, const int* in_sizes, int n_in,
                              void* d_out, int out_size, void* d_ws, size_t ws_size,
                              hipStream_t stream) {
    (void)in_sizes; (void)n_in; (void)out_size; (void)ws_size;

    const float* x   = (const float*)d_in[0];
    const int*   adj = (const int*)d_in[1];
    const float* W1  = (const float*)d_in[2];
    const float* a1  = (const float*)d_in[3];
    const float* Wo  = (const float*)d_in[4];
    const float* ao  = (const float*)d_in[5];
    float* out = (float*)d_out;

    const int* adj1 = adj + (size_t)NNODE * NNODE;  // layer 1 uses adj[1]
    const int* adj0 = adj;                           // layer 2 uses adj[0]

    // workspace layout (bytes)
    char* ws = (char*)d_ws;
    unsigned short* whT  = (unsigned short*)(ws);                       // 4 MB
    float*          f1_1 = (float*)(ws + 4194304);                      // 128 KB
    float*          f2_1 = (float*)(ws + 4194304 + 131072);             // 128 KB
    unsigned short* h1b  = (unsigned short*)(ws + 4194304 + 262144);    // 4 MB
    unsigned short* wh2T = (unsigned short*)(ws + 8650752);             // 384 KB
    float*          f12  = (float*)(ws + 8650752 + 393216);             // 16 KB
    float*          f22  = (float*)(ws + 8650752 + 393216 + 16384);     // 16 KB

    k1_wh   <<<dim3(NNODE / 16, NHID / 16, NHEADS), 32,  0, stream>>>(x, W1, whT);
    k2_f1f2 <<<dim3(NNODE / 256, NHEADS),          256,  0, stream>>>(whT, a1, f1_1, f2_1);
    k3_attn1<<<dim3(NNODE / 16),                   256,  0, stream>>>(adj1, whT, f1_1, f2_1, h1b);
    k4_wh2  <<<dim3(NNODE / 16, 3),                 32,  0, stream>>>(h1b, Wo, wh2T);
    k5_f1f2 <<<dim3(NNODE / 256),                  256,  0, stream>>>(wh2T, ao, f12, f22);
    k6_attn2<<<dim3(NNODE / 16),                    32,  0, stream>>>(adj0, wh2T, f12, f22, out);
}
